// FastSSM_51745765982535
// MI455X (gfx1250) — compile-verified, matrix operand order checked by reading, not run on hardware
//
#include <hip/hip_runtime.h>

// ---------------------------------------------------------------------------
// Mamba SSM forward (2 layers) for MI455X / gfx1250.
// bf16 activations + v_wmma_f32_16x16x32_bf16 GEMMs with LDS-staged B panel
// (async global->LDS copies), fused selective-scan kernel.
// ---------------------------------------------------------------------------

typedef __attribute__((ext_vector_type(16))) __bf16 v16bf;
typedef __attribute__((ext_vector_type(8)))  __bf16 v8bf;
typedef __attribute__((ext_vector_type(8)))  float  v8f;

union FragBF {
    v16bf f;
    v8bf  h[2];
};

#define BN_    128          // B*N sequences
#define T_     1024
#define DM_    256          // d_model
#define DI_    512          // d_inner
#define DST_   16           // d_state
#define RNK_   16           // dt_rank
#define M_     (BN_ * T_)   // 131072 rows in every GEMM

// Use the CDNA5 async global->LDS data path for B-panel staging.
#define USE_ASYNC_LDS 1

__device__ __forceinline__ void wait_asynccnt0() {
#if defined(__has_builtin) && __has_builtin(__builtin_amdgcn_s_wait_asynccnt)
    __builtin_amdgcn_s_wait_asynccnt(0);
#else
    asm volatile("s_wait_asynccnt 0x0" ::: "memory");
#endif
}

// ---------------------------------------------------------------------------
// f32 -> bf16 elementwise convert
// ---------------------------------------------------------------------------
__global__ void cvt_bf16_kernel(const float* __restrict__ in,
                                __bf16* __restrict__ out, size_t n) {
    size_t i = (size_t)blockIdx.x * blockDim.x + threadIdx.x;
    if (i < n) out[i] = (__bf16)in[i];
}

// ---------------------------------------------------------------------------
// transpose-convert: in is K x N (f32, row major), out is N x K (bf16)
// ---------------------------------------------------------------------------
__global__ void transpose_bf16_kernel(const float* __restrict__ in,
                                      __bf16* __restrict__ out,
                                      int K, int N) {
    int tid = blockIdx.x * blockDim.x + threadIdx.x;
    if (tid >= N * K) return;
    int n = tid / K;
    int k = tid % K;
    out[tid] = (__bf16)in[(size_t)k * N + n];
}

// ---------------------------------------------------------------------------
// WMMA GEMM:  C[M x N] = A[M x K] (bf16 row-major) * B[K x N]
// Bt is N x K bf16 row-major.
//   block = 256 threads (8 waves). wave w -> mtile = blockIdx.x*8 + w.
//   each wave computes NT consecutive 16-col N-tiles at blockIdx.y*NT*16.
// The B panel (NT*16 cols x K) is shared by all 8 waves: it is staged into
// LDS in fragment-swizzled order (32 lanes x 16B contiguous per fragment
// half), KCH=8 K-steps (64KB max) per pass, via async global->LDS copies.
// A fragments are double-buffered in registers.
// MODE 0: split bf16 store: cols <512 -> ob0, cols >=512 -> ob1 (stride 512)
// MODE 1: f32 store to of with row stride ostride
// MODE 2: bf16 store to ob0 with row stride ostride
// ---------------------------------------------------------------------------
template <int NT, int K, int MODE>
__global__ __launch_bounds__(256)
void gemm_bf16_kernel(const __bf16* __restrict__ A,
                      const __bf16* __restrict__ Bt,
                      __bf16* __restrict__ ob0,
                      __bf16* __restrict__ ob1,
                      float*  __restrict__ of,
                      int ostride) {
    constexpr int KS   = K / 32;                 // total K-steps
    constexpr int KCH  = (KS < 8) ? KS : 8;      // K-steps staged per pass
    constexpr int P    = KS / KCH;               // staging passes
    constexpr int NCHK = NT * KCH * 2 * 32;      // 16B chunks per pass

    __shared__ __align__(16) char smem[NT * KCH * 1024];  // NCHK * 16 bytes

    const int lane  = threadIdx.x & 31;
    const int wave  = threadIdx.x >> 5;
    const int mBase = (blockIdx.x * 8 + wave) * 16;
    const int nBase = blockIdx.y * (NT * 16);
    const int lm    = lane & 15;          // row (A) / col (B) within tile
    const int half8 = (lane >> 4) * 8;    // K-chunk select (and row-half in C)

    const __bf16* Arow = A + (size_t)(mBase + lm) * K;
    const uint32_t sbase = (uint32_t)(uintptr_t)smem;

    v8f acc[NT];
#pragma unroll
    for (int nt = 0; nt < NT; ++nt)
#pragma unroll
        for (int i = 0; i < 8; ++i) acc[nt][i] = 0.0f;

    for (int pass = 0; pass < P; ++pass) {
        __syncthreads();   // previous pass fully consumed
        // ---- stage B panel chunk into LDS (fragment-swizzled layout) ----
        for (int ci = threadIdx.x; ci < NCHK; ci += 256) {
            int L  = ci & 31;
            int c  = (ci >> 5) & 1;
            int ks = (ci >> 6) & (KCH - 1);
            int nt = ci >> 9;                     // ci / (KCH*64), KCH==8
            if (KCH != 8) nt = ci / (KCH * 64);   // (KCH<8 only when K<256)
            int n  = nBase + nt * 16 + (L & 15);
            int k  = (pass * KCH + ks) * 32 + c * 16 + (L >> 4) * 8;
            const __bf16* gp = Bt + (size_t)n * K + k;
#if USE_ASYNC_LDS
            uint32_t loff = sbase + (uint32_t)ci * 16u;
            uint64_t ga   = (uint64_t)(uintptr_t)gp;
            asm volatile("global_load_async_to_lds_b128 %0, %1, off"
                         :: "v"(loff), "v"(ga) : "memory");
#else
            *(uint4*)(smem + (size_t)ci * 16) = *(const uint4*)gp;
#endif
        }
#if USE_ASYNC_LDS
        wait_asynccnt0();
#endif
        __syncthreads();

        // ---- compute over this pass's K-steps, A double-buffered ----
        FragBF a_cur, a_nxt;
        {
            const __bf16* ap = Arow + (size_t)pass * KCH * 32 + half8;
            a_cur.h[0] = *(const v8bf*)(ap);
            a_cur.h[1] = *(const v8bf*)(ap + 16);
        }
#pragma unroll
        for (int ks = 0; ks < KCH; ++ks) {
            if (ks + 1 < KCH) {
                const __bf16* ap = Arow + (size_t)(pass * KCH + ks + 1) * 32 + half8;
                a_nxt.h[0] = *(const v8bf*)(ap);
                a_nxt.h[1] = *(const v8bf*)(ap + 16);
            } else if (pass + 1 < P) {
                const __bf16* ap = Arow + (size_t)(pass + 1) * KCH * 32 + half8;
                a_nxt.h[0] = *(const v8bf*)(ap);
                a_nxt.h[1] = *(const v8bf*)(ap + 16);
            }
#pragma unroll
            for (int nt = 0; nt < NT; ++nt) {
                const char* p = smem + ((((nt * KCH + ks) * 2) * 32 + lane) * 16);
                FragBF b;
                b.h[0] = *(const v8bf*)(p);
                b.h[1] = *(const v8bf*)(p + 512);
                acc[nt] = __builtin_amdgcn_wmma_f32_16x16x32_bf16(
                    false, a_cur.f, false, b.f, (short)0, acc[nt], false, false);
            }
            a_cur = a_nxt;
        }
    }

    // C/D layout: VGPR i, lanes 0-15 -> M=i, lanes 16-31 -> M=8+i; N = lane&15
#pragma unroll
    for (int nt = 0; nt < NT; ++nt) {
#pragma unroll
        for (int i = 0; i < 8; ++i) {
            int m = mBase + half8 + i;
            int n = nBase + nt * 16 + lm;
            float v = acc[nt][i];
            if (MODE == 0) {
                if (n < DI_) ob0[(size_t)m * DI_ + n] = (__bf16)v;
                else         ob1[(size_t)m * DI_ + (n - DI_)] = (__bf16)v;
            } else if (MODE == 1) {
                of[(size_t)m * ostride + n] = v;
            } else {
                ob0[(size_t)m * ostride + n] = (__bf16)v;
            }
        }
    }
}

// ---------------------------------------------------------------------------
// depthwise causal conv (width 4) + SiLU.  Layout (bn, t, d), d fastest.
// ---------------------------------------------------------------------------
__global__ void conv_silu_kernel(const __bf16* __restrict__ xh,
                                 __bf16* __restrict__ xc,
                                 const float* __restrict__ cw,   // [DI][4]
                                 const float* __restrict__ cb) { // [DI]
    size_t tid = (size_t)blockIdx.x * blockDim.x + threadIdx.x;
    if (tid >= (size_t)BN_ * T_ * DI_) return;
    int d   = (int)(tid % DI_);
    size_t bt = tid / DI_;
    int t   = (int)(bt % T_);
    size_t bn = bt / T_;
    float acc = cb[d];
#pragma unroll
    for (int j = 0; j < 4; ++j) {
        int tt = t - 3 + j;
        if (tt >= 0)
            acc += cw[d * 4 + j] * (float)xh[((size_t)bn * T_ + tt) * DI_ + d];
    }
    float s = acc / (1.0f + __expf(-acc));   // SiLU
    xc[tid] = (__bf16)s;
}

// ---------------------------------------------------------------------------
// fused selective scan:
//   delta = softplus(dt @ W_dt + b_dt)   (dt from x_dbl, 16 FMAs in-register)
//   h[s]  = exp(delta*A[s]) * h[s] + (delta*x) * B[s]
//   y     = sum_s h[s]*C[s];  out = (y + x*Dp) * silu(res)   (bf16, in-place)
// one block per bn (512 threads = one d each); 64-step LDS staging of x_dbl.
// ---------------------------------------------------------------------------
__global__ __launch_bounds__(512)
void scan_kernel(const float* __restrict__ xdbl,   // [bn][t][48] : dt|B|C
                 const __bf16* __restrict__ res,   // [bn][t][DI]
                 __bf16* __restrict__ xc,          // in: conv'd x, out: gated y
                 const float* __restrict__ Wdt,    // [RNK][DI]
                 const float* __restrict__ bdt,    // [DI]
                 const float* __restrict__ Alog,   // [DI][DST]
                 const float* __restrict__ Dp) {   // [DI]
    __shared__ float sh[64 * 48];
    const int d  = threadIdx.x;
    const int bn = blockIdx.x;

    float Ac[DST_], hst[DST_], wdt[RNK_];
#pragma unroll
    for (int s = 0; s < DST_; ++s) {
        Ac[s]  = -__expf(Alog[d * DST_ + s]);
        hst[s] = 0.0f;
    }
#pragma unroll
    for (int r = 0; r < RNK_; ++r) wdt[r] = Wdt[r * DI_ + d];
    const float bd  = bdt[d];
    const float dpv = Dp[d];

    for (int c = 0; c < T_ / 64; ++c) {
        __syncthreads();
        size_t base = ((size_t)bn * T_ + (size_t)c * 64) * 48;
        for (int i = threadIdx.x; i < 64 * 48; i += 512) sh[i] = xdbl[base + i];
        __syncthreads();

        for (int tt = 0; tt < 64; ++tt) {
            const float* row = &sh[tt * 48];
            float dl = bd;
#pragma unroll
            for (int r = 0; r < RNK_; ++r) dl += row[r] * wdt[r];
            float delta = (dl > 20.0f) ? dl : log1pf(__expf(dl));

            size_t idx = ((size_t)bn * T_ + (size_t)c * 64 + tt) * DI_ + d;
            float x  = (float)xc[idx];
            float dx = delta * x;
            float y  = 0.0f;
#pragma unroll
            for (int s = 0; s < DST_; ++s) {
                float dA = __expf(delta * Ac[s]);
                hst[s] = hst[s] * dA + dx * row[16 + s];
                y += hst[s] * row[32 + s];
            }
            float rv = (float)res[idx];
            float g  = rv / (1.0f + __expf(-rv));   // silu(res)
            xc[idx] = (__bf16)((y + x * dpv) * g);
        }
    }
}

// ---------------------------------------------------------------------------
// host launcher
// ---------------------------------------------------------------------------
extern "C" void kernel_launch(void* const* d_in, const int* in_sizes, int n_in,
                              void* d_out, int out_size, void* d_ws, size_t ws_size,
                              hipStream_t stream) {
    (void)in_sizes; (void)n_in; (void)out_size; (void)ws_size;
    const float* x      = (const float*)d_in[0];
    const float* W_in   = (const float*)d_in[1];
    const float* conv_w = (const float*)d_in[2];
    const float* conv_b = (const float*)d_in[3];
    const float* W_x    = (const float*)d_in[4];
    const float* W_dt   = (const float*)d_in[5];
    const float* b_dt   = (const float*)d_in[6];
    const float* A_log  = (const float*)d_in[7];
    const float* Dp     = (const float*)d_in[8];
    const float* W_out  = (const float*)d_in[9];
    float* out = (float*)d_out;

    char* ws = (char*)d_ws;
    size_t off = 0;
    auto alloc = [&](size_t bytes) -> void* {
        void* p = ws + off;
        off = (off + bytes + 255) & ~((size_t)255);
        return p;
    };
    __bf16* WinT  = (__bf16*)alloc((size_t)2 * 1024 * 256 * 2);  // [l][1024][256]
    __bf16* WxT   = (__bf16*)alloc((size_t)2 * 48 * 512 * 2);    // [l][48][512]
    __bf16* WoutT = (__bf16*)alloc((size_t)2 * 256 * 512 * 2);   // [l][256][512]
    __bf16* hbuf  = (__bf16*)alloc((size_t)M_ * DM_ * 2);        // layer input
    __bf16* xh    = (__bf16*)alloc((size_t)M_ * DI_ * 2);        // pre-conv
    __bf16* resb  = (__bf16*)alloc((size_t)M_ * DI_ * 2);        // gate branch
    __bf16* xcb   = (__bf16*)alloc((size_t)M_ * DI_ * 2);        // conv'd x / y
    float*  xdbl  = (float*) alloc((size_t)M_ * 48 * 4);         // dt|B|C

    // --- one-time weight prep: transpose + bf16-convert ---
    for (int l = 0; l < 2; ++l) {
        int n1 = 1024 * 256;
        transpose_bf16_kernel<<<(n1 + 255) / 256, 256, 0, stream>>>(
            W_in + (size_t)l * 256 * 1024, WinT + (size_t)l * 1024 * 256, 256, 1024);
        int n2 = 48 * 512;
        transpose_bf16_kernel<<<(n2 + 255) / 256, 256, 0, stream>>>(
            W_x + (size_t)l * 512 * 48, WxT + (size_t)l * 48 * 512, 512, 48);
        int n3 = 256 * 512;
        transpose_bf16_kernel<<<(n3 + 255) / 256, 256, 0, stream>>>(
            W_out + (size_t)l * 512 * 256, WoutT + (size_t)l * 256 * 512, 512, 256);
    }
    {
        size_t n = (size_t)M_ * DM_;
        cvt_bf16_kernel<<<(unsigned)((n + 255) / 256), 256, 0, stream>>>(x, hbuf, n);
    }

    const dim3 blk(256);
    for (int l = 0; l < 2; ++l) {
        // xz = h @ W_in  -> split into xh (cols 0..511) and res (cols 512..1023)
        gemm_bf16_kernel<8, DM_, 0><<<dim3(M_ / 16 / 8, 1024 / 128), blk, 0, stream>>>(
            hbuf, WinT + (size_t)l * 1024 * 256, xh, resb, nullptr, 0);

        // causal depthwise conv + SiLU
        size_t nconv = (size_t)M_ * DI_;
        conv_silu_kernel<<<(unsigned)((nconv + 255) / 256), 256, 0, stream>>>(
            xh, xcb, conv_w + (size_t)l * DI_ * 4, conv_b + (size_t)l * DI_);

        // x_dbl = xc @ W_x  (N=48, f32 out)
        gemm_bf16_kernel<3, DI_, 1><<<dim3(M_ / 16 / 8, 1), blk, 0, stream>>>(
            xcb, WxT + (size_t)l * 48 * 512, nullptr, nullptr, xdbl, 48);

        // fused delta + selective scan + skip + gate (in-place into xcb)
        scan_kernel<<<BN_, 512, 0, stream>>>(
            xdbl, resb, xcb,
            W_dt + (size_t)l * RNK_ * DI_, b_dt + (size_t)l * DI_,
            A_log + (size_t)l * DI_ * DST_, Dp + (size_t)l * DI_);

        // out = y @ W_out  (bf16 for next layer, f32 to d_out for last layer)
        if (l == 0) {
            gemm_bf16_kernel<8, DI_, 2><<<dim3(M_ / 16 / 8, 256 / 128), blk, 0, stream>>>(
                xcb, WoutT, hbuf, nullptr, nullptr, DM_);
        } else {
            gemm_bf16_kernel<8, DI_, 1><<<dim3(M_ / 16 / 8, 256 / 128), blk, 0, stream>>>(
                xcb, WoutT + (size_t)256 * 512, nullptr, nullptr, out, DM_);
        }
    }
}
